// ModifiedAttentiveFPPredictorV2_5136780886778
// MI455X (gfx1250) — matrix operand order repaired and checked by reading, not compiled, not run on hardware
//
#include <hip/hip_runtime.h>
#include <hip/hip_bf16.h>
#include <cstdint>

typedef __attribute__((ext_vector_type(16))) _Float16 v16h;
typedef __attribute__((ext_vector_type(8)))  _Float16 h8;
typedef __attribute__((ext_vector_type(8)))  float    v8f;

// ---------------- constants ----------------
#define CN   50000      // nodes
#define CE   400000     // edges
#define CNF  128        // node feat
#define CEF  64         // edge feat
#define CG   200        // hidden
#define CGP  224        // hidden padded to K%32==0
#define CKE1 192        // NF+EF
#define CG3  600        // 3*G
#define CG3P 608        // padded

enum { AM_DIRECT = 0, AM_CONCAT = 1 };
enum { ACT_NONE = 0, ACT_LEAKY = 1 };
enum { EPI_F32 = 0, EPI_F16 = 1, EPI_BOTH = 2, EPI_SCATTER = 3 };

// ---------------- helpers ----------------
__device__ __forceinline__ h8 ld8(const _Float16* p) { return *(const h8*)p; }

__device__ __forceinline__ unsigned fkey(float f) {
  unsigned b = __float_as_uint(f);
  return (b & 0x80000000u) ? ~b : (b | 0x80000000u);
}
__device__ __forceinline__ float fdec(unsigned k) {
  unsigned b = (k & 0x80000000u) ? (k ^ 0x80000000u) : ~k;
  return __uint_as_float(b);
}

template <int AMODE>
__device__ __forceinline__ h8 loadA(const _Float16* A, int lda,
                                    const _Float16* A2, int lda2,
                                    const int* gidx, int row, int k) {
  if (AMODE == AM_DIRECT) return ld8(A + (long long)row * lda + k);
  if (k < CNF) return ld8(A + (long long)gidx[row] * lda + k);
  return ld8(A2 + (long long)row * lda2 + (k - CNF));
}

// ---------------- WMMA GEMM with async-LDS weight staging ----------------
// Block = 8 waves. Each block owns (8 M-tiles) x (one N-group of 4 tiles = 64 cols).
// The 64-col weight panel (64 x Kp f16, <=28KB) is staged Global->LDS once per
// block with GLOBAL_LOAD_ASYNC_TO_LDS_B128 (ASYNCcnt); out-of-range columns are
// zero-filled in LDS so the inner K-loop is completely branch-free: each wave
// register-blocks 4 N-tiles and issues 4 unconditional v_wmma_f32_16x16x32_f16
// per K-step, reusing one A fragment from registers. Store guards are epilogue-only.
// Fragment layouts per ISA 7.12.2:
//  A: lane<16 row M=lane, K=kk+{0..7,16..23}; lane>=16 row M=lane-16, K=kk+{8..15,24..31}
//  B: lane holds column N=lane%16, K same order as A (kk+8*(lane/16)+{0..7,16..23})
//  C/D: col=lane%16, rows r+8*(lane/16)
template <int AMODE, int ACT, int EPI>
__global__ __launch_bounds__(256) void wmma_gemm(
    const _Float16* __restrict__ A, int lda,
    const _Float16* __restrict__ A2, int lda2,
    const int* __restrict__ gidx,
    const _Float16* __restrict__ Wt, int Kp,
    const float* __restrict__ bias,
    int Mtiles, int Ntiles,
    float* __restrict__ outF, int ldo, int nvalid,
    _Float16* __restrict__ outH, int ldoH,
    const int* __restrict__ dstIdx, const float* __restrict__ coef,
    float* __restrict__ acc, int ldacc)
{
  __shared__ alignas(16) _Float16 panel[64 * 224];  // max Kp = 224

  const int wave = threadIdx.x >> 5;
  const int lane = threadIdx.x & 31;
  const int Ngroups = (Ntiles + 3) >> 2;
  const int ng = blockIdx.x % Ngroups;
  const int mg = blockIdx.x / Ngroups;
  const int mt = mg * 8 + wave;
  const int nt0 = ng * 4;

  // ---- stage 64-col weight panel into LDS (async copy; zero-fill the tail) ----
  {
    const int colsValid = Ntiles * 16;          // valid rows of Wt
    const int k8cnt = Kp >> 3;                  // 16B chunks per column
    const int nchunks = Kp >> 5;                // per-thread chunks (8*Kp/256)
    for (int i = 0; i < nchunks; ++i) {
      int c = threadIdx.x + (i << 8);           // chunk id in [0, 8*Kp)
      int lcol = c / k8cnt;                     // 0..63
      int k8 = c % k8cnt;
      int gcol = ng * 64 + lcol;
      if (gcol < colsValid) {
        const _Float16* gsrc = Wt + (long long)gcol * Kp + (k8 << 3);
        unsigned dstoff = (unsigned)(uintptr_t)&panel[lcol * Kp + (k8 << 3)];
        asm volatile("global_load_async_to_lds_b128 %0, %1, off"
                     :: "v"(dstoff), "v"(gsrc) : "memory");
      } else {
        h8 z = {};
        *(h8*)&panel[lcol * Kp + (k8 << 3)] = z;  // defined zeros for padding tiles
      }
    }
    asm volatile("s_wait_asynccnt 0" ::: "memory");
    __syncthreads();
  }

  if (mt >= Mtiles) return;                     // after barrier: safe to exit

  const int hl  = lane >> 4;
  const int l16 = lane & 15;
  const int arow = mt * 16 + l16;

  v8f accv[4] = {};
  for (int kk = 0; kk < Kp; kk += 32) {
    h8 a0 = loadA<AMODE>(A, lda, A2, lda2, gidx, arow, kk + (hl ? 8 : 0));
    h8 a1 = loadA<AMODE>(A, lda, A2, lda2, gidx, arow, kk + (hl ? 24 : 16));
    v16h av;
#pragma unroll
    for (int i = 0; i < 8; ++i) { av[i] = a0[i]; av[8 + i] = a1[i]; }
    const int kb = kk + hl * 8;
#pragma unroll
    for (int j = 0; j < 4; ++j) {               // branch-free: padding cols are zero
      const _Float16* prow = &panel[(j * 16 + l16) * Kp];
      h8 b0 = ld8(prow + kb);
      h8 b1 = ld8(prow + kb + 16);
      v16h bv;
#pragma unroll
      for (int i = 0; i < 8; ++i) { bv[i] = b0[i]; bv[8 + i] = b1[i]; }
      accv[j] = __builtin_amdgcn_wmma_f32_16x16x32_f16(
          false, av, false, bv, (short)0, accv[j], false, false);
    }
  }

#pragma unroll
  for (int j = 0; j < 4; ++j) {
    if (nt0 + j >= Ntiles) continue;            // wave-uniform, epilogue only
    const int col = (nt0 + j) * 16 + l16;
    const float bb = bias[col];
#pragma unroll
    for (int r = 0; r < 8; ++r) {
      const int grow = mt * 16 + r + 8 * hl;
      float v = accv[j][r] + bb;
      if (ACT == ACT_LEAKY) v = v > 0.f ? v : 0.01f * v;
      if (EPI == EPI_F32) {
        if (col < nvalid) outF[(long long)grow * ldo + col] = v;
      } else if (EPI == EPI_F16) {
        outH[(long long)grow * ldoH + col] = (_Float16)(col < nvalid ? v : 0.f);
      } else if (EPI == EPI_BOTH) {
        if (col < nvalid) outF[(long long)grow * ldo + col] = v;
        outH[(long long)grow * ldoH + col] = (_Float16)(col < nvalid ? v : 0.f);
      } else { // EPI_SCATTER: c[dst[row]] += a[row] * v  (fused segment_sum)
        if (col < nvalid)
          atomicAdd(acc + (long long)dstIdx[grow] * ldacc + col, coef[grow] * v);
      }
    }
  }
}

// ---------------- packing / elementwise ----------------
__global__ void pack_w(const float* __restrict__ W, int K, int Nc,
                       _Float16* __restrict__ Wt, int Kp, int Np)
{
  long long total = (long long)Np * Kp;
  for (long long idx = (long long)blockIdx.x * blockDim.x + threadIdx.x;
       idx < total; idx += (long long)gridDim.x * blockDim.x) {
    int n = (int)(idx / Kp), k = (int)(idx % Kp);
    float v = (n < Nc && k < K) ? W[(long long)k * Nc + n] : 0.f;
    Wt[idx] = (_Float16)v;
  }
}

__global__ void pack_b(const float* __restrict__ b, int Nc,
                       float* __restrict__ bp, int Np)
{
  int i = blockIdx.x * blockDim.x + threadIdx.x;
  if (i < Np) bp[i] = (i < Nc) ? b[i] : 0.f;
}

// ACT: 0=none, 2=elu
template <int ACT>
__global__ void convert_pad(const float* __restrict__ src, int lds, int ncols,
                            _Float16* __restrict__ dst, int ldd, long long rows)
{
  long long total = rows * ldd;
  for (long long idx = (long long)blockIdx.x * blockDim.x + threadIdx.x;
       idx < total; idx += (long long)gridDim.x * blockDim.x) {
    long long i = idx / ldd; int j = (int)(idx % ldd);
    float v = (j < ncols) ? src[i * lds + j] : 0.f;
    if (ACT == 2) v = v > 0.f ? v : (expf(v) - 1.f);
    dst[idx] = (_Float16)v;
  }
}

__global__ void zero_f32(float* __restrict__ p, long long n) {
  for (long long i = (long long)blockIdx.x * blockDim.x + threadIdx.x;
       i < n; i += (long long)gridDim.x * blockDim.x) p[i] = 0.f;
}
__global__ void zero_u32(unsigned* __restrict__ p, long long n) {
  for (long long i = (long long)blockIdx.x * blockDim.x + threadIdx.x;
       i < n; i += (long long)gridDim.x * blockDim.x) p[i] = 0u;
}

// ---------------- edge attention logits (wave32 per edge) ----------------
__global__ __launch_bounds__(256) void edge_logits_gc(
    const float* __restrict__ hv, const _Float16* __restrict__ he, int ldh,
    const float* __restrict__ W, const float* __restrict__ b,
    const int* __restrict__ dst, float* __restrict__ out, int E)
{
  int wave = threadIdx.x >> 5, lane = threadIdx.x & 31;
  int e = blockIdx.x * 8 + wave;
  if (e >= E) return;
  long long hb = (long long)dst[e] * CG;
  long long eb = (long long)e * ldh;
  float s = 0.f;
  for (int c = lane; c < CG; c += 32)
    s += hv[hb + c] * W[c] + (float)he[eb + c] * W[CG + c];
  for (int o = 16; o > 0; o >>= 1) s += __shfl_down(s, o, 32);
  if (lane == 0) { float v = s + b[0]; out[e] = v > 0.f ? v : 0.01f * v; }
}

__global__ __launch_bounds__(256) void edge_logits_l1(
    const float* __restrict__ h0,
    const float* __restrict__ W, const float* __restrict__ b,
    const int* __restrict__ dst, const int* __restrict__ src,
    float* __restrict__ out, int E)
{
  int wave = threadIdx.x >> 5, lane = threadIdx.x & 31;
  int e = blockIdx.x * 8 + wave;
  if (e >= E) return;
  long long db = (long long)dst[e] * CG;
  long long sb = (long long)src[e] * CG;
  float s = 0.f;
  for (int c = lane; c < CG; c += 32)
    s += h0[db + c] * W[c] + h0[sb + c] * W[CG + c];
  for (int o = 16; o > 0; o >>= 1) s += __shfl_down(s, o, 32);
  if (lane == 0) { float v = s + b[0]; out[e] = v > 0.f ? v : 0.01f * v; }
}

// ---------------- edge softmax (segment max / sum via atomics) ----------------
__global__ void seg_max(const float* __restrict__ lg, const int* __restrict__ dst,
                        unsigned* __restrict__ mkey, int E)
{
  for (long long e = (long long)blockIdx.x * blockDim.x + threadIdx.x;
       e < E; e += (long long)gridDim.x * blockDim.x)
    atomicMax(&mkey[dst[e]], fkey(lg[e]));
}
__global__ void seg_expsum(const float* __restrict__ lg, const int* __restrict__ dst,
                           const unsigned* __restrict__ mkey,
                           float* __restrict__ ebuf, float* __restrict__ s, int E)
{
  for (long long e = (long long)blockIdx.x * blockDim.x + threadIdx.x;
       e < E; e += (long long)gridDim.x * blockDim.x) {
    int d = dst[e];
    float v = expf(lg[e] - fdec(mkey[d]));
    ebuf[e] = v;
    atomicAdd(&s[d], v);
  }
}
__global__ void seg_norm(const float* __restrict__ ebuf, const int* __restrict__ dst,
                         const float* __restrict__ s, float* __restrict__ a,
                         float plus, int E)
{
  for (long long e = (long long)blockIdx.x * blockDim.x + threadIdx.x;
       e < E; e += (long long)gridDim.x * blockDim.x)
    a[e] = ebuf[e] / s[dst[e]] + plus;
}

// ---------------- GRU (biases folded into gx/gh by GEMM epilogue) + relu ----------------
__global__ void gru_relu(const float* __restrict__ gx, const float* __restrict__ gh,
                         const float* __restrict__ h, float* __restrict__ out,
                         long long total /* N*G */)
{
  for (long long idx = (long long)blockIdx.x * blockDim.x + threadIdx.x;
       idx < total; idx += (long long)gridDim.x * blockDim.x) {
    long long i = idx / CG; int g = (int)(idx % CG);
    const float* gxr = gx + i * CG3;
    const float* ghr = gh + i * CG3;
    float r  = 1.f / (1.f + expf(-(gxr[g] + ghr[g])));
    float z  = 1.f / (1.f + expf(-(gxr[CG + g] + ghr[CG + g])));
    float nn = tanhf(gxr[2 * CG + g] + r * ghr[2 * CG + g]);
    float hp = (1.f - z) * nn + z * h[idx];
    out[idx] = hp > 0.f ? hp : 0.f;
  }
}

// ---------------- layer-1 weighted scatter: c2[dst] += a2[e]*hvp[src] ----------------
__global__ void scatter_pn(const float* __restrict__ hvp, const float* __restrict__ a2,
                           const int* __restrict__ src, const int* __restrict__ dst,
                           float* __restrict__ c2, long long total /* E*G */)
{
  for (long long idx = (long long)blockIdx.x * blockDim.x + threadIdx.x;
       idx < total; idx += (long long)gridDim.x * blockDim.x) {
    long long e = idx / CG; int g = (int)(idx % CG);
    atomicAdd(&c2[(long long)dst[e] * CG + g],
              a2[e] * hvp[(long long)src[e] * CG + g]);
  }
}

// ---------------- batchnorm ----------------
__global__ __launch_bounds__(256) void bn_stats(const float* __restrict__ h1,
                                                float* __restrict__ mu,
                                                float* __restrict__ var)
{
  __shared__ float ss[256], sq[256];
  int g = blockIdx.x;  // one block per column
  float s = 0.f, q = 0.f;
  for (int i = threadIdx.x; i < CN; i += blockDim.x) {
    float v = h1[(long long)i * CG + g];
    s += v; q += v * v;
  }
  ss[threadIdx.x] = s; sq[threadIdx.x] = q;
  __syncthreads();
  for (int o = 128; o > 0; o >>= 1) {
    if (threadIdx.x < o) { ss[threadIdx.x] += ss[threadIdx.x + o];
                           sq[threadIdx.x] += sq[threadIdx.x + o]; }
    __syncthreads();
  }
  if (threadIdx.x == 0) {
    float m = ss[0] / (float)CN;
    mu[g] = m;
    var[g] = sq[0] / (float)CN - m * m;
  }
}

__global__ void bn_out(const float* __restrict__ h0, const float* __restrict__ h1,
                       const float* __restrict__ mu, const float* __restrict__ var,
                       const float* __restrict__ gam, const float* __restrict__ bet,
                       float* __restrict__ out, long long total)
{
  for (long long idx = (long long)blockIdx.x * blockDim.x + threadIdx.x;
       idx < total; idx += (long long)gridDim.x * blockDim.x) {
    int g = (int)(idx % CG);
    float v = (h1[idx] - mu[g]) * rsqrtf(var[g] + 1e-5f);
    out[idx] = h0[idx] + gam[g] * v + bet[g];
  }
}

// ---------------- host ----------------
static inline int cdiv_i(long long a, long long b) { return (int)((a + b - 1) / b); }
static inline int gemm_grid(int Mtiles, int Ntiles) {
  return ((Mtiles + 7) / 8) * ((Ntiles + 3) / 4);
}

extern "C" void kernel_launch(void* const* d_in, const int* in_sizes, int n_in,
                              void* d_out, int out_size, void* d_ws, size_t ws_size,
                              hipStream_t stream)
{
  (void)in_sizes; (void)n_in; (void)out_size; (void)ws_size;

  const float* node_feats = (const float*)d_in[0];
  const float* edge_feats = (const float*)d_in[1];
  const float* gc_node_W = (const float*)d_in[2];
  const float* gc_node_b = (const float*)d_in[3];
  const float* gc_e1_W = (const float*)d_in[4];
  const float* gc_e1_b = (const float*)d_in[5];
  const float* gc_e2_W = (const float*)d_in[6];
  const float* gc_e2_b = (const float*)d_in[7];
  const float* gc_et_W = (const float*)d_in[8];
  const float* gc_et_b = (const float*)d_in[9];
  const float* gc_gru_Wx = (const float*)d_in[10];
  const float* gc_gru_Wh = (const float*)d_in[11];
  const float* gc_gru_bx = (const float*)d_in[12];
  const float* gc_gru_bh = (const float*)d_in[13];
  const float* l1_e_W = (const float*)d_in[14];
  const float* l1_e_b = (const float*)d_in[15];
  const float* l1_pn_W = (const float*)d_in[16];
  const float* l1_pn_b = (const float*)d_in[17];
  const float* l1_gru_Wx = (const float*)d_in[18];
  const float* l1_gru_Wh = (const float*)d_in[19];
  const float* l1_gru_bx = (const float*)d_in[20];
  const float* l1_gru_bh = (const float*)d_in[21];
  const float* bn_g = (const float*)d_in[22];
  const float* bn_b = (const float*)d_in[23];
  const int* src = (const int*)d_in[24];
  const int* dst = (const int*)d_in[25];
  float* out = (float*)d_out;

  // ---- workspace carve (256B aligned) ----
  char* base = (char*)d_ws;
  size_t off = 0;
  auto alloc = [&](size_t bytes) -> char* {
    char* p = base + off;
    off += (bytes + 255) & ~(size_t)255;
    return p;
  };
  _Float16* nf16 = (_Float16*)alloc((size_t)CN * CNF * 2);
  _Float16* ef16 = (_Float16*)alloc((size_t)CE * CEF * 2);
  _Float16* WtN  = (_Float16*)alloc((size_t)CGP * CNF * 2);  float* bN  = (float*)alloc(CGP * 4);
  _Float16* WtE1 = (_Float16*)alloc((size_t)CGP * CKE1 * 2); float* bE1 = (float*)alloc(CGP * 4);
  _Float16* WtET = (_Float16*)alloc((size_t)CGP * CGP * 2);  float* bET = (float*)alloc(CGP * 4);
  _Float16* WtGX = (_Float16*)alloc((size_t)CG3P * CGP * 2); float* bGX = (float*)alloc(CG3P * 4);
  _Float16* WtGH = (_Float16*)alloc((size_t)CG3P * CGP * 2); float* bGH = (float*)alloc(CG3P * 4);
  _Float16* WtPN = (_Float16*)alloc((size_t)CGP * CGP * 2);  float* bPN = (float*)alloc(CGP * 4);
  _Float16* WtLX = (_Float16*)alloc((size_t)CG3P * CGP * 2); float* bLX = (float*)alloc(CG3P * 4);
  _Float16* WtLH = (_Float16*)alloc((size_t)CG3P * CGP * 2); float* bLH = (float*)alloc(CG3P * 4);
  float*    hv   = (float*)alloc((size_t)CN * CG * 4);
  _Float16* hv16 = (_Float16*)alloc((size_t)CN * CGP * 2);
  _Float16* he16 = (_Float16*)alloc((size_t)CE * CGP * 2);
  float*    lg   = (float*)alloc((size_t)CE * 4);
  float*    ebuf = (float*)alloc((size_t)CE * 4);
  float*    abuf = (float*)alloc((size_t)CE * 4);
  unsigned* mkey = (unsigned*)alloc((size_t)CN * 4);
  float*    sbuf = (float*)alloc((size_t)CN * 4);
  float*    cbuf = (float*)alloc((size_t)CN * CG * 4);  // also reused as c2
  _Float16* x16  = (_Float16*)alloc((size_t)CN * CGP * 2);
  float*    gxb  = (float*)alloc((size_t)CN * CG3 * 4);
  float*    ghb  = (float*)alloc((size_t)CN * CG3 * 4);
  float*    h0   = (float*)alloc((size_t)CN * CG * 4);
  _Float16* h016 = (_Float16*)alloc((size_t)CN * CGP * 2);
  float*    hvp  = (float*)alloc((size_t)CN * CG * 4);
  float*    h1   = (float*)alloc((size_t)CN * CG * 4);
  float*    muB  = (float*)alloc(CGP * 4);
  float*    varB = (float*)alloc(CGP * 4);

  const int T = 256;
  const int MtN = CN / 16;     // 3125 node tiles
  const int MtE = CE / 16;     // 25000 edge tiles
  const int NtG14 = CGP / 16;  // 14 tiles -> cols 0..223 (padded f16 writes)
  const int NtG13 = 13;        // cols 0..207 (f32 guarded writes)
  const int Nt38  = CG3P / 16; // 38 tiles for 600-col GEMMs

  // ---- 0) pack weights (transpose + f16 + zero-pad) ----
  pack_w<<<cdiv_i((long long)CGP * CNF, T), T, 0, stream>>>(gc_node_W, CNF, CG, WtN, CNF, CGP);
  pack_b<<<1, CGP, 0, stream>>>(gc_node_b, CG, bN, CGP);
  pack_w<<<cdiv_i((long long)CGP * CKE1, T), T, 0, stream>>>(gc_e1_W, CKE1, CG, WtE1, CKE1, CGP);
  pack_b<<<1, CGP, 0, stream>>>(gc_e1_b, CG, bE1, CGP);
  pack_w<<<cdiv_i((long long)CGP * CGP, T), T, 0, stream>>>(gc_et_W, CG, CG, WtET, CGP, CGP);
  pack_b<<<1, CGP, 0, stream>>>(gc_et_b, CG, bET, CGP);
  pack_w<<<cdiv_i((long long)CG3P * CGP, T), T, 0, stream>>>(gc_gru_Wx, CG, CG3, WtGX, CGP, CG3P);
  pack_b<<<cdiv_i(CG3P, T), T, 0, stream>>>(gc_gru_bx, CG3, bGX, CG3P);
  pack_w<<<cdiv_i((long long)CG3P * CGP, T), T, 0, stream>>>(gc_gru_Wh, CG, CG3, WtGH, CGP, CG3P);
  pack_b<<<cdiv_i(CG3P, T), T, 0, stream>>>(gc_gru_bh, CG3, bGH, CG3P);
  pack_w<<<cdiv_i((long long)CGP * CGP, T), T, 0, stream>>>(l1_pn_W, CG, CG, WtPN, CGP, CGP);
  pack_b<<<1, CGP, 0, stream>>>(l1_pn_b, CG, bPN, CGP);
  pack_w<<<cdiv_i((long long)CG3P * CGP, T), T, 0, stream>>>(l1_gru_Wx, CG, CG3, WtLX, CGP, CG3P);
  pack_b<<<cdiv_i(CG3P, T), T, 0, stream>>>(l1_gru_bx, CG3, bLX, CG3P);
  pack_w<<<cdiv_i((long long)CG3P * CGP, T), T, 0, stream>>>(l1_gru_Wh, CG, CG3, WtLH, CGP, CG3P);
  pack_b<<<cdiv_i(CG3P, T), T, 0, stream>>>(l1_gru_bh, CG3, bLH, CG3P);

  // ---- 1) feature conversion to f16 ----
  convert_pad<0><<<cdiv_i((long long)CN * CNF, T), T, 0, stream>>>(node_feats, CNF, CNF, nf16, CNF, CN);
  convert_pad<0><<<cdiv_i((long long)CE * CEF, T), T, 0, stream>>>(edge_feats, CEF, CEF, ef16, CEF, CE);

  // ---- 2) hv_new = leaky(node_feats @ W + b)  (f32 + padded f16) ----
  wmma_gemm<AM_DIRECT, ACT_LEAKY, EPI_BOTH>
      <<<gemm_grid(MtN, NtG14), T, 0, stream>>>(
      nf16, CNF, nullptr, 0, nullptr, WtN, CNF, bN, MtN, NtG14,
      hv, CG, CG, hv16, CGP, nullptr, nullptr, nullptr, 0);

  // ---- 3) he1 = leaky(concat(nf[src], ef) @ W + b)  (padded f16) ----
  wmma_gemm<AM_CONCAT, ACT_LEAKY, EPI_F16>
      <<<gemm_grid(MtE, NtG14), T, 0, stream>>>(
      nf16, CNF, ef16, CEF, src, WtE1, CKE1, bE1, MtE, NtG14,
      nullptr, 0, CG, he16, CGP, nullptr, nullptr, nullptr, 0);

  // ---- 4) logits + edge softmax ----
  edge_logits_gc<<<cdiv_i(CE, 8), T, 0, stream>>>(hv, he16, CGP, gc_e2_W, gc_e2_b, dst, lg, CE);
  zero_u32<<<cdiv_i(CN, T), T, 0, stream>>>(mkey, CN);
  zero_f32<<<cdiv_i(CN, T), T, 0, stream>>>(sbuf, CN);
  seg_max<<<cdiv_i(CE, T), T, 0, stream>>>(lg, dst, mkey, CE);
  seg_expsum<<<cdiv_i(CE, T), T, 0, stream>>>(lg, dst, mkey, ebuf, sbuf, CE);
  seg_norm<<<cdiv_i(CE, T), T, 0, stream>>>(ebuf, dst, sbuf, abuf, 0.0f, CE);

  // ---- 5) c = segment_sum(a * (he1 @ et_W + b)) : fused WMMA + atomic scatter ----
  zero_f32<<<cdiv_i((long long)CN * CG, T), T, 0, stream>>>(cbuf, (long long)CN * CG);
  wmma_gemm<AM_DIRECT, ACT_NONE, EPI_SCATTER>
      <<<gemm_grid(MtE, NtG13), T, 0, stream>>>(
      he16, CGP, nullptr, 0, nullptr, WtET, CGP, bET, MtE, NtG13,
      nullptr, 0, CG, nullptr, 0, dst, abuf, cbuf, CG);

  // ---- 6) GRU(elu(c), hv_new) -> h0 ----
  convert_pad<2><<<cdiv_i((long long)CN * CGP, T), T, 0, stream>>>(cbuf, CG, CG, x16, CGP, CN);
  wmma_gemm<AM_DIRECT, ACT_NONE, EPI_F32>
      <<<gemm_grid(MtN, Nt38), T, 0, stream>>>(
      x16, CGP, nullptr, 0, nullptr, WtGX, CGP, bGX, MtN, Nt38,
      gxb, CG3, CG3, nullptr, 0, nullptr, nullptr, nullptr, 0);
  wmma_gemm<AM_DIRECT, ACT_NONE, EPI_F32>
      <<<gemm_grid(MtN, Nt38), T, 0, stream>>>(
      hv16, CGP, nullptr, 0, nullptr, WtGH, CGP, bGH, MtN, Nt38,
      ghb, CG3, CG3, nullptr, 0, nullptr, nullptr, nullptr, 0);
  gru_relu<<<cdiv_i((long long)CN * CG, T), T, 0, stream>>>(gxb, ghb, hv, h0, (long long)CN * CG);
  convert_pad<0><<<cdiv_i((long long)CN * CGP, T), T, 0, stream>>>(h0, CG, CG, h016, CGP, CN);

  // ---- 7) layer-1 attention: a2 = softmax(leaky([h0[dst],h0[src]]@W+b)) + 1 ----
  edge_logits_l1<<<cdiv_i(CE, 8), T, 0, stream>>>(h0, l1_e_W, l1_e_b, dst, src, lg, CE);
  zero_u32<<<cdiv_i(CN, T), T, 0, stream>>>(mkey, CN);
  zero_f32<<<cdiv_i(CN, T), T, 0, stream>>>(sbuf, CN);
  seg_max<<<cdiv_i(CE, T), T, 0, stream>>>(lg, dst, mkey, CE);
  seg_expsum<<<cdiv_i(CE, T), T, 0, stream>>>(lg, dst, mkey, ebuf, sbuf, CE);
  seg_norm<<<cdiv_i(CE, T), T, 0, stream>>>(ebuf, dst, sbuf, abuf, 1.0f, CE);

  // ---- 8) hvp = h0 @ pn_W + b ; c2 = segment_sum(hvp[src] * a2) ----
  wmma_gemm<AM_DIRECT, ACT_NONE, EPI_F32>
      <<<gemm_grid(MtN, NtG13), T, 0, stream>>>(
      h016, CGP, nullptr, 0, nullptr, WtPN, CGP, bPN, MtN, NtG13,
      hvp, CG, CG, nullptr, 0, nullptr, nullptr, nullptr, 0);
  zero_f32<<<cdiv_i((long long)CN * CG, T), T, 0, stream>>>(cbuf, (long long)CN * CG);
  scatter_pn<<<cdiv_i((long long)CE * CG, T), T, 0, stream>>>(hvp, abuf, src, dst, cbuf, (long long)CE * CG);

  // ---- 9) GRU(elu(c2), h0) -> h1 ----
  convert_pad<2><<<cdiv_i((long long)CN * CGP, T), T, 0, stream>>>(cbuf, CG, CG, x16, CGP, CN);
  wmma_gemm<AM_DIRECT, ACT_NONE, EPI_F32>
      <<<gemm_grid(MtN, Nt38), T, 0, stream>>>(
      x16, CGP, nullptr, 0, nullptr, WtLX, CGP, bLX, MtN, Nt38,
      gxb, CG3, CG3, nullptr, 0, nullptr, nullptr, nullptr, 0);
  wmma_gemm<AM_DIRECT, ACT_NONE, EPI_F32>
      <<<gemm_grid(MtN, Nt38), T, 0, stream>>>(
      h016, CGP, nullptr, 0, nullptr, WtLH, CGP, bLH, MtN, Nt38,
      ghb, CG3, CG3, nullptr, 0, nullptr, nullptr, nullptr, 0);
  gru_relu<<<cdiv_i((long long)CN * CG, T), T, 0, stream>>>(gxb, ghb, h0, h1, (long long)CN * CG);

  // ---- 10) batchnorm(h1) ; out = h0 + bn(h1) ----
  bn_stats<<<CG, T, 0, stream>>>(h1, muB, varB);
  bn_out<<<cdiv_i((long long)CN * CG, T), T, 0, stream>>>(h0, h1, muB, varB, bn_g, bn_b, out, (long long)CN * CG);
}
